// EncoderDecoder_46145128628690
// MI455X (gfx1250) — compile-verified, matrix operand order checked by reading, not run on hardware
//
#include <hip/hip_runtime.h>
#include <hip/hip_bf16.h>
#include <math.h>

#define V_SIZE 32000
#define EMB    512
#define HDIM   1024
#define BATCH  32
#define L_INS  128
#define L_OUTS 64
#define G3H    (3 * HDIM)
#define START_TOK 1

typedef __attribute__((ext_vector_type(16))) __bf16 v16bf;
typedef __attribute__((ext_vector_type(8)))  float  v8f;

// ---------------- helpers ----------------

__device__ inline __bf16 f2bf(float x) {
    union { float f; unsigned u; } v; v.f = x;
    unsigned r = v.u + 0x7FFFu + ((v.u >> 16) & 1u);   // round-to-nearest-even
    unsigned short h = (unsigned short)(r >> 16);
    return __builtin_bit_cast(__bf16, h);
}

__device__ inline float sigmoidf(float x) { return 1.0f / (1.0f + expf(-x)); }

// ---------------- conversions / init ----------------

__global__ void k_f32_to_bf16(const float* __restrict__ src, __bf16* __restrict__ dst, int n) {
    int i = blockIdx.x * 256 + threadIdx.x;
    if (i < n) dst[i] = f2bf(src[i]);
}

__global__ void k_zero_f32(float* __restrict__ p, int n) {
    int i = blockIdx.x * 256 + threadIdx.x;
    if (i < n) p[i] = 0.0f;
}

__global__ void k_zero_bf16(__bf16* __restrict__ p, int n) {
    int i = blockIdx.x * 256 + threadIdx.x;
    if (i < n) p[i] = __builtin_bit_cast(__bf16, (unsigned short)0);
}

__global__ void k_fill_int(int* __restrict__ p, int n, int v) {
    int i = blockIdx.x * 256 + threadIdx.x;
    if (i < n) p[i] = v;
}

// ---------------- embedding gathers (produce bf16 GEMM operands) ----------------

__global__ void k_emb_enc(const int* __restrict__ inputs, const float* __restrict__ emb,
                          __bf16* __restrict__ xEmb) {
    int idx = blockIdx.x * 256 + threadIdx.x;          // B*L_IN*EMB = 2,097,152
    int row = idx >> 9;                                // (b*L_IN + t)
    int e   = idx & 511;
    xEmb[idx] = f2bf(emb[(size_t)inputs[row] * EMB + e]);
}

__global__ void k_emb_dec(const int* __restrict__ tok, const float* __restrict__ emb,
                          __bf16* __restrict__ gin) {
    int idx = blockIdx.x * 256 + threadIdx.x;          // B*EMB = 16384
    int b = idx >> 9, e = idx & 511;
    gin[b * (HDIM + EMB) + HDIM + e] = f2bf(emb[(size_t)tok[b] * EMB + e]);
}

// ---------------- WMMA GEMM: C[M,N] = A[M,K](bf16) @ W[N,K](bf16)^T + bias ----------------
// grid = (N/128, M/32), block = 256 (8 waves). Each wave: one 32x16 output tile,
// B-fragment (weights) loaded once per k-step and reused for both M subtiles.
// A fragment (16-bit A 16x32): lane<16 -> row=lane,    K = k0+0..15
//                              lane>=16 -> row=lane-16, K = k0+16..31
// B fragment (16-bit B 32x16): lane<16 -> col=lane, K = k0+0..15 = W[n, k0..k0+15] contiguous
// C (f32 16x16): lane<16: N=lane, M=r ; lane>=16: N=lane-16, M=8+r

__global__ __launch_bounds__(256) void k_gemm_bf16(
    const __bf16* __restrict__ A, int lda,
    const __bf16* __restrict__ W,
    const float* __restrict__ bias,
    float* __restrict__ C, long long ldc,
    int K)
{
    int wave   = threadIdx.x >> 5;
    int lane   = threadIdx.x & 31;
    int lane15 = lane & 15;
    int khalf  = (lane >> 4) << 4;                  // 0 or 16
    int ncol   = (blockIdx.x * 8 + wave) * 16 + lane15;
    int mbase  = blockIdx.y * 32;

    v8f acc0 = {};
    v8f acc1 = {};

    const __bf16* a0base = A + (size_t)(mbase + lane15) * lda + khalf;
    const __bf16* a1base = A + (size_t)(mbase + 16 + lane15) * lda + khalf;
    const __bf16* wbase  = W + (size_t)ncol * K + khalf;

    for (int k0 = 0; k0 < K; k0 += 32) {
        v16bf bfrag  = *(const v16bf*)(wbase  + k0);   // 32B contiguous, aligned
        v16bf afrag0 = *(const v16bf*)(a0base + k0);
        v16bf afrag1 = *(const v16bf*)(a1base + k0);
        acc0 = __builtin_amdgcn_wmma_f32_16x16x32_bf16(false, afrag0, false, bfrag,
                                                       (short)0, acc0, false, false);
        acc1 = __builtin_amdgcn_wmma_f32_16x16x32_bf16(false, afrag1, false, bfrag,
                                                       (short)0, acc1, false, false);
    }

    float bval = bias ? bias[ncol] : 0.0f;
    int mlo = (lane < 16) ? 0 : 8;
#pragma unroll
    for (int r = 0; r < 8; ++r) {
        int m = mlo + r;
        C[(long long)(mbase + m)      * ldc + ncol] = acc0[r] + bval;
        C[(long long)(mbase + 16 + m) * ldc + ncol] = acc1[r] + bval;
    }
}

// ---------------- GRU pointwise gates (emits f32 h + bf16 shadow for GEMMs) ----------------

__global__ void k_gru(const float* __restrict__ gi, const float* __restrict__ gh,
                      const float* __restrict__ h,
                      float* __restrict__ hnext, __bf16* __restrict__ hnextBf,
                      float* __restrict__ encStore /* +t*HDIM, or null */,
                      __bf16* __restrict__ encStoreBf /* +t*HDIM, or null */)
{
    int idx = blockIdx.x * 256 + threadIdx.x;          // B*HDIM = 32768
    int b = idx >> 10, j = idx & 1023;
    const float* gib = gi + b * G3H;
    const float* ghb = gh + b * G3H;
    float r  = sigmoidf(gib[j] + ghb[j]);
    float z  = sigmoidf(gib[HDIM + j] + ghb[HDIM + j]);
    float nn = tanhf(gib[2 * HDIM + j] + r * ghb[2 * HDIM + j]);
    float h2 = (1.0f - z) * nn + z * h[idx];
    hnext[idx]   = h2;
    hnextBf[idx] = f2bf(h2);
    if (encStore) {
        encStore[(size_t)b * (L_INS * HDIM) + j]   = h2;
        encStoreBf[(size_t)b * (L_INS * HDIM) + j] = f2bf(h2);
    }
}

// ---------------- attention score: Vw . tanh(enc_proj + q) + Vb ----------------

__global__ __launch_bounds__(256) void k_score(
    const float* __restrict__ encProj, const float* __restrict__ q,
    const float* __restrict__ Vw, const float* __restrict__ Vb,
    float* __restrict__ score)
{
    int w    = blockIdx.x * 8 + (threadIdx.x >> 5);    // wave id over B*L_IN = 4096
    int lane = threadIdx.x & 31;
    int b = w >> 7, l = w & 127;
    const float* ep = encProj + (size_t)(b * L_INS + l) * HDIM;
    const float* qb = q + b * HDIM;
    float s = 0.0f;
    for (int h = lane; h < HDIM; h += 32)
        s += Vw[h] * tanhf(ep[h] + qb[h]);
    for (int off = 16; off > 0; off >>= 1) s += __shfl_down(s, off, 32);
    if (lane == 0) score[b * L_INS + l] = s + Vb[0];
}

// ---------------- softmax over L_IN, writes attn workspace + output slice ----------------

__global__ void k_softmax(const float* __restrict__ score, float* __restrict__ attn,
                          float* __restrict__ outAttn /* d_out attn base + t*L_IN */)
{
    __shared__ float sm[L_INS];
    int b = blockIdx.x, l = threadIdx.x;
    float v = score[b * L_INS + l];
    sm[l] = v; __syncthreads();
    for (int off = 64; off > 0; off >>= 1) {
        if (l < off) sm[l] = fmaxf(sm[l], sm[l + off]);
        __syncthreads();
    }
    float mx = sm[0]; __syncthreads();
    float e = expf(v - mx);
    sm[l] = e; __syncthreads();
    for (int off = 64; off > 0; off >>= 1) {
        if (l < off) sm[l] += sm[l + off];
        __syncthreads();
    }
    float a = e / sm[0];
    attn[b * L_INS + l] = a;
    outAttn[(size_t)b * (L_OUTS * L_INS) + l] = a;
}

// ------- context: ctx[b,h] = sum_l attn[b,l]*enc_out[b,l,h] -> ginBf[:, :H] (bf16) -------

__global__ void k_ctx(const float* __restrict__ attn, const float* __restrict__ encOut,
                      __bf16* __restrict__ gin)
{
    int idx = blockIdx.x * 256 + threadIdx.x;          // B*HDIM = 32768
    int b = idx >> 10, h = idx & 1023;
    const float* eo = encOut + (size_t)b * L_INS * HDIM + h;
    const float* ab = attn + b * L_INS;
    float s = 0.0f;
    for (int l = 0; l < L_INS; ++l) s += ab[l] * eo[(size_t)l * HDIM];
    gin[b * (HDIM + EMB) + h] = f2bf(s);
}

// ---------------- argmax over vocab (first-max tie-break) ----------------

__global__ void k_argmax(const float* __restrict__ logits /* d_out + t*V */,
                         int* __restrict__ tok)
{
    __shared__ float sv[256];
    __shared__ int   si[256];
    int b = blockIdx.x, tid = threadIdx.x;
    const float* row = logits + (size_t)b * ((size_t)L_OUTS * V_SIZE);
    float best = -INFINITY; int bi = 0;
    for (int i = tid; i < V_SIZE; i += 256) {
        float v = row[i];
        if (v > best) { best = v; bi = i; }
    }
    sv[tid] = best; si[tid] = bi; __syncthreads();
    for (int off = 128; off > 0; off >>= 1) {
        if (tid < off) {
            float v2 = sv[tid + off]; int i2 = si[tid + off];
            if (v2 > sv[tid] || (v2 == sv[tid] && i2 < si[tid])) { sv[tid] = v2; si[tid] = i2; }
        }
        __syncthreads();
    }
    if (tid == 0) tok[b] = si[0];
}

// ---------------- host orchestration ----------------

extern "C" void kernel_launch(void* const* d_in, const int* in_sizes, int n_in,
                              void* d_out, int out_size, void* d_ws, size_t ws_size,
                              hipStream_t stream)
{
    const int*   inputs  = (const int*)  d_in[0];
    const float* emb_enc = (const float*)d_in[1];
    const float* enc_wih = (const float*)d_in[2];
    const float* enc_whh = (const float*)d_in[3];
    const float* enc_bih = (const float*)d_in[4];
    const float* enc_bhh = (const float*)d_in[5];
    const float* emb_dec = (const float*)d_in[6];
    const float* dec_wih = (const float*)d_in[7];
    const float* dec_whh = (const float*)d_in[8];
    const float* dec_bih = (const float*)d_in[9];
    const float* dec_bhh = (const float*)d_in[10];
    const float* W1      = (const float*)d_in[11];
    const float* b1      = (const float*)d_in[12];
    const float* W2      = (const float*)d_in[13];
    const float* b2      = (const float*)d_in[14];
    const float* Vw      = (const float*)d_in[15];
    const float* Vb      = (const float*)d_in[16];
    const float* fc_w    = (const float*)d_in[17];
    const float* fc_b    = (const float*)d_in[18];
    float* out = (float*)d_out;

    char* ws = (char*)d_ws;
    size_t off = 0;
    auto alloc = [&](size_t bytes) -> void* {
        void* p = ws + off;
        off += (bytes + 255) & ~(size_t)255;
        return p;
    };

    // bf16 weight copies (L2-resident working set; fc_w drops 131MB->65.5MB)
    __bf16* encWihBf = (__bf16*)alloc((size_t)G3H * EMB * 2);
    __bf16* encWhhBf = (__bf16*)alloc((size_t)G3H * HDIM * 2);
    __bf16* decWihBf = (__bf16*)alloc((size_t)G3H * (HDIM + EMB) * 2);
    __bf16* decWhhBf = (__bf16*)alloc((size_t)G3H * HDIM * 2);
    __bf16* W1Bf     = (__bf16*)alloc((size_t)HDIM * HDIM * 2);
    __bf16* W2Bf     = (__bf16*)alloc((size_t)HDIM * HDIM * 2);
    __bf16* fcBf     = (__bf16*)alloc((size_t)V_SIZE * HDIM * 2);

    // bf16 activation buffers (GEMM A operands)
    __bf16* xEmbBf   = (__bf16*)alloc((size_t)BATCH * L_INS * EMB * 2);
    __bf16* encOutBf = (__bf16*)alloc((size_t)BATCH * L_INS * HDIM * 2);
    __bf16* hABf     = (__bf16*)alloc((size_t)BATCH * HDIM * 2);
    __bf16* hBBf     = (__bf16*)alloc((size_t)BATCH * HDIM * 2);
    __bf16* ginBf    = (__bf16*)alloc((size_t)BATCH * (HDIM + EMB) * 2);

    // f32 intermediates
    float* encOut  = (float*)alloc((size_t)BATCH * L_INS * HDIM * 4);
    float* encProj = (float*)alloc((size_t)BATCH * L_INS * HDIM * 4);
    float* hA      = (float*)alloc((size_t)BATCH * HDIM * 4);
    float* hB      = (float*)alloc((size_t)BATCH * HDIM * 4);
    float* gi      = (float*)alloc((size_t)BATCH * G3H * 4);
    float* gh      = (float*)alloc((size_t)BATCH * G3H * 4);
    float* q       = (float*)alloc((size_t)BATCH * HDIM * 4);
    float* score   = (float*)alloc((size_t)BATCH * L_INS * 4);
    float* attn    = (float*)alloc((size_t)BATCH * L_INS * 4);
    int*   tok     = (int*)  alloc((size_t)BATCH * 4);

    auto conv = [&](const float* s, __bf16* d, int n) {
        k_f32_to_bf16<<<(n + 255) / 256, 256, 0, stream>>>(s, d, n);
    };
    auto gemm = [&](const __bf16* A, int lda, const __bf16* Wm, const float* bias,
                    float* C, long long ldc, int M, int N, int K) {
        dim3 g(N / 128, M / 32);
        k_gemm_bf16<<<g, 256, 0, stream>>>(A, lda, Wm, bias, C, ldc, K);
    };

    // --- one-time (per launch) prep ---
    conv(enc_wih, encWihBf, G3H * EMB);
    conv(enc_whh, encWhhBf, G3H * HDIM);
    conv(dec_wih, decWihBf, G3H * (HDIM + EMB));
    conv(dec_whh, decWhhBf, G3H * HDIM);
    conv(W1,      W1Bf,     HDIM * HDIM);
    conv(W2,      W2Bf,     HDIM * HDIM);
    conv(fc_w,    fcBf,     V_SIZE * HDIM);

    {
        int n = BATCH * L_INS * EMB;
        k_emb_enc<<<(n + 255) / 256, 256, 0, stream>>>(inputs, emb_enc, xEmbBf);
    }
    k_zero_f32<<<(BATCH * HDIM + 255) / 256, 256, 0, stream>>>(hA, BATCH * HDIM);
    k_zero_bf16<<<(BATCH * HDIM + 255) / 256, 256, 0, stream>>>(hABf, BATCH * HDIM);
    k_fill_int<<<1, 32, 0, stream>>>(tok, BATCH, START_TOK);

    float*  hcur   = hA;
    float*  hnxt   = hB;
    __bf16* hcurBf = hABf;
    __bf16* hnxtBf = hBBf;

    // --- encoder: 128 sequential GRU steps ---
    for (int t = 0; t < L_INS; ++t) {
        gemm(xEmbBf + (size_t)t * EMB, L_INS * EMB, encWihBf, enc_bih, gi, G3H, BATCH, G3H, EMB);
        gemm(hcurBf, HDIM, encWhhBf, enc_bhh, gh, G3H, BATCH, G3H, HDIM);
        k_gru<<<BATCH * HDIM / 256, 256, 0, stream>>>(gi, gh, hcur, hnxt, hnxtBf,
                                                      encOut + (size_t)t * HDIM,
                                                      encOutBf + (size_t)t * HDIM);
        float*  t1 = hcur;   hcur   = hnxt;   hnxt   = t1;
        __bf16* t2 = hcurBf; hcurBf = hnxtBf; hnxtBf = t2;
    }

    // --- enc_proj = enc_output @ W1^T + b1  (M = 4096) ---
    gemm(encOutBf, HDIM, W1Bf, b1, encProj, HDIM, BATCH * L_INS, HDIM, HDIM);

    // --- decoder: 64 sequential attention + GRU + vocab-projection steps ---
    const long long ldcOut = (long long)L_OUTS * V_SIZE;
    float* outAttnBase = out + (size_t)BATCH * L_OUTS * V_SIZE;
    for (int t = 0; t < L_OUTS; ++t) {
        gemm(hcurBf, HDIM, W2Bf, b2, q, HDIM, BATCH, HDIM, HDIM);
        k_score<<<(BATCH * L_INS) / 8, 256, 0, stream>>>(encProj, q, Vw, Vb, score);
        k_softmax<<<BATCH, L_INS, 0, stream>>>(score, attn, outAttnBase + (size_t)t * L_INS);
        k_ctx<<<BATCH * HDIM / 256, 256, 0, stream>>>(attn, encOut, ginBf);
        k_emb_dec<<<(BATCH * EMB) / 256, 256, 0, stream>>>(tok, emb_dec, ginBf);
        gemm(ginBf, HDIM + EMB, decWihBf, dec_bih, gi, G3H, BATCH, G3H, HDIM + EMB);
        gemm(hcurBf, HDIM, decWhhBf, dec_bhh, gh, G3H, BATCH, G3H, HDIM);
        k_gru<<<BATCH * HDIM / 256, 256, 0, stream>>>(gi, gh, hcur, hnxt, hnxtBf,
                                                      (float*)nullptr, (__bf16*)nullptr);
        // logits written directly into output slice outputs[:, t, :]
        gemm(hnxtBf, HDIM, fcBf, fc_b, out + (size_t)t * V_SIZE, ldcOut, BATCH, V_SIZE, HDIM);
        k_argmax<<<BATCH, 256, 0, stream>>>(out + (size_t)t * V_SIZE, tok);
        float*  t1 = hcur;   hcur   = hnxt;   hnxt   = t1;
        __bf16* t2 = hcurBf; hcurBf = hnxtBf; hnxtBf = t2;
    }
}
